// ChainCRF_3118146257273
// MI455X (gfx1250) — matrix operand order repaired and checked by reading, not compiled, not run on hardware
//
#include <hip/hip_runtime.h>

// ---------------------------------------------------------------------------
// ChainCRF loss on gfx1250 (MI455X).
// Phase 1: pack x -> f16, build combined weight W_c^T (f16) + combined bias.
// Phase 2: WMMA f16 GEMM  E[m,n] = (x[m,:]·W_c[:,n] + bc[n]) * mask[m]
//          (m = b*T+t, n = i*L+j), v_wmma_f32_16x16x32_f16, f32 accum.
//          Wave tile 32(M)x64(N): 8 WMMAs per K-step off 12 b128 loads.
// Phase 3: per-batch forward recursion (logsumexp over previous label) on E.
// ---------------------------------------------------------------------------

typedef __attribute__((ext_vector_type(16))) _Float16 v16h;
typedef __attribute__((ext_vector_type(8)))  float    v8f;

#define L_LAB 51
#define LL    2601            // L*L
#define K_IN  256
#define B_SZ  64
#define T_SZ  256
#define BT    (B_SZ * T_SZ)   // 16384 rows
#define NPAD  2624            // LL padded up to 41*64 (multiple of 64)
#define NBLK  (NPAD / 64)     // 41 N-blocks of 64 columns

// ---------------- pack x (f32 -> f16, row-major [BT, 256]) -----------------
__global__ void pack_x_kernel(const float* __restrict__ x,
                              _Float16* __restrict__ xh, int n) {
    int i = blockIdx.x * blockDim.x + threadIdx.x;
    if (i < n) xh[i] = (_Float16)x[i];
}

// ------- build W_c^T [NPAD, 256] f16 (K contiguous) + combined bias --------
// W_c[k, i*L+j] = trans_W[k, i*L+j] + state_W[k, j];  bc[n] = trans_b + state_b
__global__ void pack_w_kernel(const float* __restrict__ tW,
                              const float* __restrict__ sW,
                              const float* __restrict__ tb,
                              const float* __restrict__ sb,
                              _Float16* __restrict__ wct,
                              float* __restrict__ bc) {
    int tid = blockIdx.x * 256 + threadIdx.x;
    int n = tid >> 8;          // output column (0..NPAD-1)
    int k = tid & 255;         // input dim
    if (n >= NPAD) return;
    float v = 0.0f;
    if (n < LL) {
        int j = n % L_LAB;
        v = tW[(size_t)k * LL + n] + sW[(size_t)k * L_LAB + j];
    }
    wct[(size_t)n * K_IN + k] = (_Float16)v;
    if (k == 0) {
        float b = 0.0f;
        if (n < LL) b = tb[n] + sb[n % L_LAB];
        bc[n] = b;
    }
}

// ----------------------------- WMMA GEMM -----------------------------------
// Block = 256 threads = 8 waves. Block tile 256(M) x 64(N); each wave a
// 32(M) x 64(N) strip: 2 M-tiles x 4 N-tiles = 8 accumulators, each B
// fragment shared by 2 WMMAs, each A fragment by 4. K-loop of 8 x 32.
union AFrag {
    v16h v;
    struct { uint4 lo, hi; } q;
};

__global__ __launch_bounds__(256)
void energy_gemm_kernel(const _Float16* __restrict__ A,   // [BT, 256] f16
                        const _Float16* __restrict__ Bw,  // [NPAD, 256] f16 (W_c^T)
                        const float* __restrict__ bc,     // [NPAD]
                        const float* __restrict__ mask,   // [BT]
                        float* __restrict__ E) {          // [BT, NPAD] f32
    const int bx   = blockIdx.x % NBLK;   // N block
    const int by   = blockIdx.x / NBLK;   // M block
    const int wave = threadIdx.x >> 5;
    const int lane = threadIdx.x & 31;
    const int r    = lane & 15;
    const int hi   = lane >> 4;

    const int mBase = by * 256 + wave * 32;   // two 16-row M tiles per wave
    const int nBase = bx * 64;

    v8f acc[2][4] = {};

    const _Float16* aRow0 = A + (size_t)(mBase + r) * K_IN;
    const _Float16* aRow1 = A + (size_t)(mBase + 16 + r) * K_IN;

    #pragma unroll
    for (int kk = 0; kk < K_IN; kk += 32) {
        // A 16x32 f16 fragments: lane holds row r,
        // K = hi*16 + [0..15] (matches B)  -> two 16B loads each.
        AFrag a0, a1;
        a0.q.lo = *(const uint4*)(aRow0 + kk + hi * 16);
        a0.q.hi = *(const uint4*)(aRow0 + kk + hi * 16 + 8);
        a1.q.lo = *(const uint4*)(aRow1 + kk + hi * 16);
        a1.q.hi = *(const uint4*)(aRow1 + kk + hi * 16 + 8);

        #pragma unroll
        for (int t = 0; t < 4; ++t) {
            // B 32x16 f16 fragment: lane holds column n, K = hi*16 + [0..15]
            // contiguous in W_c^T -> one aligned 32B load, used by 2 WMMAs.
            const _Float16* bp =
                Bw + (size_t)(nBase + t * 16 + r) * K_IN + kk + hi * 16;
            v16h bf = *(const v16h*)bp;
            acc[0][t] = __builtin_amdgcn_wmma_f32_16x16x32_f16(
                false, a0.v, false, bf, (short)0, acc[0][t], false, false);
            acc[1][t] = __builtin_amdgcn_wmma_f32_16x16x32_f16(
                false, a1.v, false, bf, (short)0, acc[1][t], false, false);
        }
    }

    // C/D layout: col = nBase + t*16 + r ; row = mBase + u*16 + hi*8 + v
    float mk[2][8];
    #pragma unroll
    for (int u = 0; u < 2; ++u)
        #pragma unroll
        for (int v = 0; v < 8; ++v)
            mk[u][v] = mask[mBase + u * 16 + hi * 8 + v];

    #pragma unroll
    for (int t = 0; t < 4; ++t) {
        const int n = nBase + t * 16 + r;
        const float bias = bc[n];
        #pragma unroll
        for (int u = 0; u < 2; ++u) {
            #pragma unroll
            for (int v = 0; v < 8; ++v) {
                const int m = mBase + u * 16 + hi * 8 + v;
                E[(size_t)m * NPAD + n] = (acc[u][t][v] + bias) * mk[u][v];
            }
        }
    }
}

// ------------------------ CRF forward recursion ----------------------------
// One 64-thread block per batch element. part[j] lives in LDS; each t-step
// stages the 2601-float energy slice into LDS, then 51 threads each do an
// online-softmax logsumexp over the 51 previous labels.
__global__ __launch_bounds__(64)
void crf_forward_kernel(const float* __restrict__ E,
                        const float* __restrict__ mask,
                        const int* __restrict__ tgt,
                        float* __restrict__ out) {
    __shared__ float part[64];
    __shared__ float eL[LL];
    __shared__ float tgtE;
    __shared__ int   prevS;

    const int b = blockIdx.x;
    const int j = threadIdx.x;
    const size_t row0 = (size_t)(b * T_SZ) * NPAD;

    // t = 0: start from pad label (row L-1 of the transition slab)
    if (j < L_LAB) part[j] = E[row0 + (L_LAB - 1) * L_LAB + j];
    if (j == 0) {
        int t0 = tgt[b * T_SZ];
        tgtE  = E[row0 + (L_LAB - 1) * L_LAB + t0];
        prevS = t0;
    }
    __syncthreads();

    for (int t = 1; t < T_SZ; ++t) {
        const size_t row = row0 + (size_t)t * NPAD;
        for (int idx = j; idx < LL; idx += 64) eL[idx] = E[row + idx];
        __syncthreads();

        float pn = 0.0f;
        if (j < L_LAB) {
            float mx = -1e30f, s = 0.0f;
            for (int i = 0; i < L_LAB; ++i) {
                float v = eL[i * L_LAB + j] + part[i];
                if (v > mx) { s = s * __expf(mx - v) + 1.0f; mx = v; }
                else        { s += __expf(v - mx); }
            }
            pn = mx + __logf(s);
        }
        __syncthreads();

        if (j < L_LAB) {
            float m = mask[b * T_SZ + t];
            part[j] = part[j] + (pn - part[j]) * m;
        }
        if (j == 0) {
            int tg = tgt[b * T_SZ + t];
            tgtE += eL[prevS * L_LAB + tg];   // E already mask-scaled
            prevS = tg;
        }
        __syncthreads();
    }

    if (j == 0) {
        float mx = part[0];
        for (int i = 1; i < L_LAB; ++i) mx = fmaxf(mx, part[i]);
        float s = 0.0f;
        for (int i = 0; i < L_LAB; ++i) s += __expf(part[i] - mx);
        out[b] = mx + __logf(s) - tgtE;
    }
}

// ---------------------------------------------------------------------------
extern "C" void kernel_launch(void* const* d_in, const int* in_sizes, int n_in,
                              void* d_out, int out_size, void* d_ws, size_t ws_size,
                              hipStream_t stream) {
    const float* x      = (const float*)d_in[0];  // [B,T,256]
    const float* mask   = (const float*)d_in[1];  // [B,T]
    const int*   target = (const int*)  d_in[2];  // [B,T]
    const float* sW     = (const float*)d_in[3];  // [256, 51]
    const float* sb     = (const float*)d_in[4];  // [51]
    const float* tW     = (const float*)d_in[5];  // [256, 2601]
    const float* tb     = (const float*)d_in[6];  // [2601]
    float* out          = (float*)d_out;          // [B]

    // Workspace layout (~182 MB):
    //   xh  : BT*256 f16      (8.39 MB)
    //   wct : NPAD*256 f16    (1.34 MB)
    //   bc  : NPAD f32        (10.5 KB)
    //   E   : BT*NPAD f32     (172 MB)
    char* ws = (char*)d_ws;
    _Float16* xh = (_Float16*)ws;
    size_t off = (size_t)BT * K_IN * sizeof(_Float16);
    _Float16* wct = (_Float16*)(ws + off);
    off += (size_t)NPAD * K_IN * sizeof(_Float16);
    float* bc = (float*)(ws + off);
    off += (size_t)NPAD * sizeof(float);
    off = (off + 255) & ~(size_t)255;
    float* E = (float*)(ws + off);

    pack_x_kernel<<<(BT * K_IN + 255) / 256, 256, 0, stream>>>(x, xh, BT * K_IN);
    pack_w_kernel<<<NPAD, 256, 0, stream>>>(tW, sW, tb, sb, wct, bc);
    energy_gemm_kernel<<<(BT / 256) * NBLK, 256, 0, stream>>>(xh, wct, bc, mask, E);
    crf_forward_kernel<<<B_SZ, 64, 0, stream>>>(E, mask, target, out);
}
